// EGATDecoder_89644557402626
// MI455X (gfx1250) — compile-verified, matrix operand order checked by reading, not run on hardware
//
#include <hip/hip_runtime.h>

#define N_NODES 20000
#define E0_EDGES 640000
#define E_EDGES (E0_EDGES + N_NODES)   // 660000, divisible by 4

typedef _Float16 half_t;
typedef half_t v16h __attribute__((ext_vector_type(16)));
typedef half_t v8h  __attribute__((ext_vector_type(8)));
typedef float  v8f  __attribute__((ext_vector_type(8)));

__device__ __forceinline__ unsigned enc_f32(float f) {
    unsigned u = __float_as_uint(f);
    return (u & 0x80000000u) ? ~u : (u ^ 0x80000000u);
}
__device__ __forceinline__ float dec_f32(unsigned e) {
    unsigned u = (e & 0x80000000u) ? (e ^ 0x80000000u) : ~e;
    return __uint_as_float(u);
}

// Build a WMMA 16-bit operand fragment: two contiguous 8-half runs at
// lds[base] and lds[base+16] (per ISA 7.12.2 K-layout for 16x16x32).
__device__ __forceinline__ v16h frag_ld(const half_t* lds, int base) {
    v8h lo = *(const v8h*)(lds + base);
    v8h hi = *(const v8h*)(lds + base + 16);
    return __builtin_shufflevector(lo, hi, 0, 1, 2, 3, 4, 5, 6, 7,
                                   8, 9, 10, 11, 12, 13, 14, 15);
}

// C[M x 256] = A[M x K] * W[K x 256] (+ optional bias[256]); f16 WMMA, f32 accum.
// Block: 256 threads = 8 waves; block tile 64(M) x 256(N); wave tile 32 x 64.
__global__ __launch_bounds__(256) void gemm_wmma(const float* __restrict__ A,
        const float* __restrict__ W, const float* __restrict__ bias,
        float* __restrict__ C, int M, int K) {
    __shared__ __align__(16) half_t lA[64 * 32];    // [row][k]
    __shared__ __align__(16) half_t lBt[256 * 32];  // [col][k] (transposed)
    const int tid   = threadIdx.x;
    const int lane  = tid & 31;
    const int wave  = tid >> 5;
    const int wm    = wave >> 2;      // 0..1  (rows 32*wm)
    const int wn    = wave & 3;       // 0..3  (cols 64*wn)
    const int laneN = lane & 15;
    const int khalf = lane >> 4;
    const int rowBase = blockIdx.x * 64;

    v8f acc[2][4] = {};

    const int arow  = tid >> 2;                 // 0..63
    const int akoff = (tid & 3) * 8;            // 0,8,16,24
    const int arowc = min(rowBase + arow, M - 1);

    for (int k0 = 0; k0 < K; k0 += 32) {
        // A tile 64x32: 8 halves per thread, unconditional clamped loads.
        {
            const float* ap = A + (size_t)arowc * K + k0 + akoff;
            float4 f0 = *(const float4*)(ap);
            float4 f1 = *(const float4*)(ap + 4);
            half_t* o = lA + arow * 32 + akoff;
            o[0] = (half_t)f0.x; o[1] = (half_t)f0.y;
            o[2] = (half_t)f0.z; o[3] = (half_t)f0.w;
            o[4] = (half_t)f1.x; o[5] = (half_t)f1.y;
            o[6] = (half_t)f1.z; o[7] = (half_t)f1.w;
        }
        // B tile 32x256 stored transposed: thread owns one output column.
        {
            half_t tmp[32];
            #pragma unroll
            for (int r = 0; r < 32; ++r)
                tmp[r] = (half_t)W[(size_t)(k0 + r) * 256 + tid];
            half_t* o = lBt + tid * 32;
            #pragma unroll
            for (int r = 0; r < 4; ++r)
                *(v8h*)(o + r * 8) = *(const v8h*)(tmp + r * 8);
        }
        __syncthreads();

        v16h af[2];
        #pragma unroll
        for (int mi = 0; mi < 2; ++mi)
            af[mi] = frag_ld(lA, (wm * 32 + mi * 16 + laneN) * 32 + khalf * 8);
        #pragma unroll
        for (int ni = 0; ni < 4; ++ni) {
            v16h bf = frag_ld(lBt, (wn * 64 + ni * 16 + laneN) * 32 + khalf * 8);
            #pragma unroll
            for (int mi = 0; mi < 2; ++mi)
                acc[mi][ni] = __builtin_amdgcn_wmma_f32_16x16x32_f16(
                    false, af[mi], false, bf, (short)0, acc[mi][ni], false, false);
        }
        __syncthreads();
    }

    #pragma unroll
    for (int mi = 0; mi < 2; ++mi) {
        #pragma unroll
        for (int ni = 0; ni < 4; ++ni) {
            int col = wn * 64 + ni * 16 + laneN;
            float bv = bias ? bias[col] : 0.0f;
            #pragma unroll
            for (int r = 0; r < 8; ++r) {
                int rowIn = (lane < 16) ? r : (r + 8);
                int row = rowBase + wm * 32 + mi * 16 + rowIn;
                if (row < M) C[(size_t)row * 256 + col] = acc[mi][ni][r] + bv;
            }
        }
    }
}

__global__ void init_layer(float* __restrict__ hout, float* __restrict__ den,
                           unsigned* __restrict__ nmax) {
    int i = blockIdx.x * blockDim.x + threadIdx.x;
    if (i < N_NODES * 256) hout[i] = 0.0f;
    if (i < N_NODES * 4) { den[i] = 0.0f; nmax[i] = 0x007FFFFFu; }  // enc(-inf)
}

// 64 threads per edge (f = 0..63), 4 edges per 256-thread block.
__global__ __launch_bounds__(256) void edge_assemble(
        const float* __restrict__ fni, const float* __restrict__ fnj,
        const float* __restrict__ fij, int fij_rows,
        const float* __restrict__ bias, const float* __restrict__ attn,
        const int* __restrict__ src, const int* __restrict__ dst,
        float* __restrict__ enext, float* __restrict__ logits,
        unsigned* __restrict__ nmax) {
    __shared__ float sums[4][4];
    const int t = threadIdx.x;
    if (t < 16) sums[t >> 2][t & 3] = 0.0f;
    __syncthreads();

    const int le = t >> 6;
    const int f  = t & 63;
    const int eid = blockIdx.x * 4 + le;
    int s, d;
    if (eid < E0_EDGES) { s = src[eid]; d = dst[eid]; }
    else                { s = eid - E0_EDGES; d = s; }

    float l[4];
    float vsum = 0.0f;
    #pragma unroll
    for (int h = 0; h < 4; ++h) {
        int idx = h * 64 + f;
        float v = fni[(size_t)s * 256 + idx] + fnj[(size_t)d * 256 + idx] + bias[idx];
        if (eid < fij_rows) v += fij[(size_t)eid * 256 + idx];
        v = v > 0.0f ? v : 0.01f * v;            // leaky_relu
        vsum += v;
        l[h] = v * attn[idx];
    }
    enext[(size_t)eid * 64 + f] = vsum * 0.25f;  // head mean -> next-layer e

    #pragma unroll
    for (int h = 0; h < 4; ++h) {
        #pragma unroll
        for (int off = 16; off > 0; off >>= 1)
            l[h] += __shfl_xor(l[h], off, 32);
    }
    if ((t & 31) == 0) {
        #pragma unroll
        for (int h = 0; h < 4; ++h) atomicAdd(&sums[le][h], l[h]);
    }
    __syncthreads();

    if (t < 16) {
        int le2 = t >> 2, h = t & 3;
        int e2 = blockIdx.x * 4 + le2;
        float lv = sums[le2][h];
        logits[(size_t)e2 * 4 + h] = lv;
        int d2 = (e2 < E0_EDGES) ? dst[e2] : (e2 - E0_EDGES);
        atomicMax(&nmax[d2 * 4 + h], enc_f32(lv));
    }
}

__global__ void edge_exp(const int* __restrict__ dst, float* __restrict__ logits,
                         const unsigned* __restrict__ nmax, float* __restrict__ den) {
    int i = blockIdx.x * blockDim.x + threadIdx.x;
    if (i >= E_EDGES * 4) return;
    int eid = i >> 2, h = i & 3;
    int d = (eid < E0_EDGES) ? dst[eid] : (eid - E0_EDGES);
    float m = dec_f32(nmax[d * 4 + h]);
    float ex = __expf(logits[i] - m);
    logits[i] = ex;                               // reuse buffer for exp values
    atomicAdd(&den[d * 4 + h], ex);
}

__global__ void edge_scatter(const int* __restrict__ src, const int* __restrict__ dst,
        const float* __restrict__ hn, const float* __restrict__ ex,
        const float* __restrict__ den, float* __restrict__ hout) {
    long long i = (long long)blockIdx.x * 256 + threadIdx.x;
    if (i >= (long long)E_EDGES * 256) return;
    int eid = (int)(i >> 8), j = (int)(i & 255), h = j >> 6;
    int s, d;
    if (eid < E0_EDGES) { s = src[eid]; d = dst[eid]; }
    else                { s = eid - E0_EDGES; d = s; }
    float a = ex[(size_t)eid * 4 + h] / den[d * 4 + h];
    atomicAdd(&hout[(size_t)d * 256 + j], hn[(size_t)s * 256 + j] * a);
}

__global__ void node_finalize(const float* __restrict__ hout, float* __restrict__ xout) {
    int i = blockIdx.x * blockDim.x + threadIdx.x;
    if (i >= N_NODES * 64) return;
    int n = i >> 6, f = i & 63;
    const float* p = hout + (size_t)n * 256 + f;
    float m = 0.25f * (p[0] + p[64] + p[128] + p[192]);   // head mean
    xout[i] = m > 0.0f ? m : (__expf(m) - 1.0f);          // ELU
}

extern "C" void kernel_launch(void* const* d_in, const int* in_sizes, int n_in,
                              void* d_out, int out_size, void* d_ws, size_t ws_size,
                              hipStream_t stream) {
    (void)in_sizes; (void)n_in; (void)out_size; (void)ws_size;
    const float* x0  = (const float*)d_in[0];
    const float* ef  = (const float*)d_in[1];
    const int*   src = (const int*)d_in[2];
    const int*   dst = (const int*)d_in[3];

    float* ws = (float*)d_ws;
    size_t off = 0;
    float* fni  = ws + off; off += (size_t)N_NODES * 256;
    float* fnj  = ws + off; off += (size_t)N_NODES * 256;
    float* hn   = ws + off; off += (size_t)N_NODES * 256;
    float* hout = ws + off; off += (size_t)N_NODES * 256;
    float* xbuf = ws + off; off += (size_t)N_NODES * 64;
    float* fij  = ws + off; off += (size_t)E_EDGES * 256;
    float* eA   = ws + off; off += (size_t)E_EDGES * 64;
    float* eB   = ws + off; off += (size_t)E_EDGES * 64;
    float* lg   = ws + off; off += (size_t)E_EDGES * 4;
    float* den  = ws + off; off += (size_t)N_NODES * 4;
    unsigned* nmax = (unsigned*)(ws + off); off += (size_t)N_NODES * 4;

    dim3 blk(256);
    for (int layer = 0; layer < 3; ++layer) {
        int pb = 4 + layer * 7;
        const float* W_node = (const float*)d_in[pb + 0];
        const float* b_node = (const float*)d_in[pb + 1];
        const float* W_ni   = (const float*)d_in[pb + 2];
        const float* W_nj   = (const float*)d_in[pb + 3];
        const float* W_fij  = (const float*)d_in[pb + 4];
        const float* attn   = (const float*)d_in[pb + 5];
        const float* bias   = (const float*)d_in[pb + 6];

        const float* xin = (layer == 0) ? x0 : xbuf;
        int K = (layer == 0) ? 128 : 64;
        const float* ein = (layer == 0) ? ef : ((layer == 1) ? eA : eB);
        int Me = (layer == 0) ? E0_EDGES : E_EDGES;   // layer-0 self-loop e is zero
        float* eout = (layer == 1) ? eB : eA;

        dim3 gN((N_NODES + 63) / 64);
        gemm_wmma<<<gN, blk, 0, stream>>>(xin, W_ni,   nullptr, fni, N_NODES, K);
        gemm_wmma<<<gN, blk, 0, stream>>>(xin, W_nj,   nullptr, fnj, N_NODES, K);
        gemm_wmma<<<gN, blk, 0, stream>>>(xin, W_node, b_node,  hn,  N_NODES, K);
        dim3 gE((Me + 63) / 64);
        gemm_wmma<<<gE, blk, 0, stream>>>(ein, W_fij,  nullptr, fij, Me, K);

        init_layer<<<(N_NODES * 256 + 255) / 256, blk, 0, stream>>>(hout, den, nmax);
        edge_assemble<<<E_EDGES / 4, blk, 0, stream>>>(fni, fnj, fij, Me, bias, attn,
                                                       src, dst, eout, lg, nmax);
        edge_exp<<<(E_EDGES * 4 + 255) / 256, blk, 0, stream>>>(dst, lg, nmax, den);
        edge_scatter<<<(int)(((long long)E_EDGES * 256 + 255) / 256), blk, 0, stream>>>(
            src, dst, hn, lg, den, hout);
        float* xo = (layer == 2) ? (float*)d_out : xbuf;
        node_finalize<<<(N_NODES * 64 + 255) / 256, blk, 0, stream>>>(hout, xo);
    }
}
